// GATModule_11605001634177
// MI455X (gfx1250) — matrix operand order repaired and checked by reading, not compiled, not run on hardware
//
#include <hip/hip_runtime.h>

typedef __attribute__((ext_vector_type(16))) _Float16 v16h;
typedef __attribute__((ext_vector_type(8)))  float    v8f;

static constexpr int NNODES = 20000;
static constexpr int NEDGES = 320000;
static constexpr int TEDGES = NEDGES + NNODES;   // with self-loops appended

// ---------------- utility kernels ----------------
__global__ void fill_f32(float* __restrict__ p, float v, long n){
  long i = (long)blockIdx.x*blockDim.x + threadIdx.x;
  long s = (long)gridDim.x*blockDim.x;
  for(; i < n; i += s) p[i] = v;
}

__global__ void cvt_f32_f16(const float* __restrict__ in, _Float16* __restrict__ o, long n){
  long i = (long)blockIdx.x*blockDim.x + threadIdx.x;
  long s = (long)gridDim.x*blockDim.x;
  for(; i < n; i += s) o[i] = (_Float16)in[i];
}

// in: K x N (row-major f32) -> out: N x K (row-major f16)   (weight pre-transpose)
__global__ void cvt_t_f16(const float* __restrict__ in, _Float16* __restrict__ o, int K, int N){
  long i = (long)blockIdx.x*blockDim.x + threadIdx.x;
  long tot = (long)K*N;
  long s = (long)gridDim.x*blockDim.x;
  for(; i < tot; i += s){
    int n = (int)(i / K);
    int k = (int)(i % K);
    o[i] = (_Float16)in[(size_t)k*N + n];
  }
}

__device__ __forceinline__ float wave_sum(float v){
#pragma unroll
  for(int o2 = 16; o2 > 0; o2 >>= 1) v += __shfl_xor(v, o2, 32);
  return v;
}

__device__ __forceinline__ void atomicMaxF(float* addr, float val){
  unsigned int* ai = (unsigned int*)addr;
  unsigned int old = *ai;
  while(__uint_as_float(old) < val){
    unsigned int assumed = old;
    old = atomicCAS(ai, assumed, __float_as_uint(val));
    if(old == assumed) break;
  }
}

// ---------------- WMMA GEMM (B pre-transposed, no LDS, SW-pipelined) ----------------
// out = [relu]( A(f16, MxK row-major) * B + bias ), B given as BT (NxK row-major f16).
// Requires K%64==0, N%64==0; M arbitrary. Block = 256 threads (8 waves),
// tile = 128(M) x 64(N); each wave: 16 rows x 4 WMMA tiles along N.
// Two-stage double buffer over K so loads overlap the matrix pipe.
struct VFrag { int4 q[2]; };

__global__ __launch_bounds__(256) void wmma_gemm(
    const _Float16* __restrict__ A, const _Float16* __restrict__ BT,
    const float* __restrict__ bias,
    float* __restrict__ outF, _Float16* __restrict__ outH,
    int M, int N, int K, int relu)
{
  const int lane = threadIdx.x & 31;
  const int wv   = threadIdx.x >> 5;
  const int l16  = lane & 15;
  const int hi   = lane >> 4;
  const int mBase = blockIdx.x*128 + wv*16;
  const int nBase = blockIdx.y*64;

  int row = mBase + l16; if(row >= M) row = M - 1;   // clamp: keeps EXEC full for WMMA
  const _Float16* aRow = A + (size_t)row*K + hi*8;

  const _Float16* bRow[4];
#pragma unroll
  for(int s = 0; s < 4; s++)
    bRow[s] = BT + (size_t)(nBase + s*16 + l16)*K + hi*8;

  v8f acc[4] = {};
  VFrag a0, a1, b0[4], b1[4];

#define LOAD_FRAG(k0, a, b)                                   \
  do {                                                        \
    (a).q[0] = *(const int4*)(aRow + (k0));                   \
    (a).q[1] = *(const int4*)(aRow + (k0) + 16);              \
    _Pragma("unroll")                                         \
    for(int s = 0; s < 4; s++){                               \
      (b)[s].q[0] = *(const int4*)(bRow[s] + (k0));           \
      (b)[s].q[1] = *(const int4*)(bRow[s] + (k0) + 16);      \
    }                                                         \
  } while(0)

#define WMMA4(a, b)                                           \
  do {                                                        \
    _Pragma("unroll")                                         \
    for(int s = 0; s < 4; s++){                               \
      acc[s] = __builtin_amdgcn_wmma_f32_16x16x32_f16(        \
          false, *(const v16h*)&(a), false,                   \
          *(const v16h*)&(b)[s], (short)0, acc[s],            \
          false, false);                                      \
    }                                                         \
  } while(0)

  LOAD_FRAG(0,  a0, b0);
  LOAD_FRAG(32, a1, b1);

  int k0 = 0;
#pragma unroll 1
  for(; k0 < K - 64; k0 += 64){
    WMMA4(a0, b0);
    LOAD_FRAG(k0 + 64, a0, b0);    // prefetch while stage-1 computes
    WMMA4(a1, b1);
    LOAD_FRAG(k0 + 96, a1, b1);    // prefetch while next stage-0 computes
  }
  WMMA4(a0, b0);                   // peeled last 64
  WMMA4(a1, b1);

#undef LOAD_FRAG
#undef WMMA4

#pragma unroll
  for(int s = 0; s < 4; s++){
    const int n = nBase + s*16 + l16;
    const float bv = bias ? bias[n] : 0.f;
#pragma unroll
    for(int v = 0; v < 8; v++){
      const int m = mBase + hi*8 + v;          // C/D layout: VGPR v -> M = v (+8 for hi)
      if(m < M){
        float val = acc[s][v] + bv;
        if(relu) val = fmaxf(val, 0.f);
        if(outF) outF[(size_t)m*N + n] = val;
        if(outH) outH[(size_t)m*N + n] = (_Float16)val;
      }
    }
  }
}

// ---------------- GAT attention kernels ----------------
__global__ void gat_scores(const float* __restrict__ xh, const float* __restrict__ as,
                           const float* __restrict__ ad, float* __restrict__ s_out,
                           float* __restrict__ d_out_, int n)
{
  int wid  = (blockIdx.x*blockDim.x + threadIdx.x) >> 5;
  int lane = threadIdx.x & 31;
  if(wid >= n) return;
  const float* xr = xh + (size_t)wid*512;
  float s[4] = {0,0,0,0}, d[4] = {0,0,0,0};
  for(int c = lane; c < 128; c += 32){
#pragma unroll
    for(int h = 0; h < 4; h++){
      float xv = xr[h*128 + c];
      s[h] += xv*as[h*128 + c];
      d[h] += xv*ad[h*128 + c];
    }
  }
#pragma unroll
  for(int h = 0; h < 4; h++){ s[h] = wave_sum(s[h]); d[h] = wave_sum(d[h]); }
  if(lane == 0){
#pragma unroll
    for(int h = 0; h < 4; h++){ s_out[wid*4+h] = s[h]; d_out_[wid*4+h] = d[h]; }
  }
}

__device__ __forceinline__ void edge_sd(const int* ei, int e, int& src, int& dst){
  if(e < NEDGES){ src = ei[e]; dst = ei[NEDGES + e]; }
  else          { src = dst = e - NEDGES; }          // appended self-loops
}

__global__ void edge_max(const int* __restrict__ ei, const float* __restrict__ a_s,
                         const float* __restrict__ a_d, float* __restrict__ amax)
{
  int e = blockIdx.x*blockDim.x + threadIdx.x;
  if(e >= TEDGES) return;
  int src, dst; edge_sd(ei, e, src, dst);
#pragma unroll
  for(int h = 0; h < 4; h++){
    float a = a_s[src*4+h] + a_d[dst*4+h];
    a = a > 0.f ? a : 0.2f*a;                        // leaky_relu(.,0.2)
    atomicMaxF(&amax[dst*4+h], a);
  }
}

__global__ void edge_expsum(const int* __restrict__ ei, const float* __restrict__ a_s,
                            const float* __restrict__ a_d, const float* __restrict__ amax,
                            float* __restrict__ ex, float* __restrict__ den)
{
  int e = blockIdx.x*blockDim.x + threadIdx.x;
  if(e >= TEDGES) return;
  int src, dst; edge_sd(ei, e, src, dst);
#pragma unroll
  for(int h = 0; h < 4; h++){
    float a = a_s[src*4+h] + a_d[dst*4+h];
    a = a > 0.f ? a : 0.2f*a;
    float v = expf(a - amax[dst*4+h]);
    ex[(size_t)e*4 + h] = v;
    atomicAdd(&den[dst*4+h], v);
  }
}

// one 128-thread block per edge: thread t handles channels [t*4, t*4+4)
__global__ void edge_aggregate(const int* __restrict__ ei, const float* __restrict__ xh,
                               const float* __restrict__ ex, const float* __restrict__ den,
                               float* __restrict__ agg)
{
  int e = blockIdx.x;
  int t = threadIdx.x;
  int src, dst; edge_sd(ei, e, src, dst);
  int h = t >> 5;                                    // 128 ch per head
  float attn = ex[(size_t)e*4 + h] / (den[dst*4+h] + 1e-16f);
  float4 xv = *(const float4*)(xh + (size_t)src*512 + t*4);
  float* o = agg + (size_t)dst*512 + t*4;            // agg (41MB) is L2-resident
  atomicAdd(o+0, xv.x*attn);
  atomicAdd(o+1, xv.y*attn);
  atomicAdd(o+2, xv.z*attn);
  atomicAdd(o+3, xv.w*attn);
}

__global__ void bias_relu_cvt(const float* __restrict__ in, const float* __restrict__ b,
                              _Float16* __restrict__ o, long n, int mask){
  long i = (long)blockIdx.x*blockDim.x + threadIdx.x;
  long s = (long)gridDim.x*blockDim.x;
  for(; i < n; i += s) o[i] = (_Float16)fmaxf(in[i] + b[i & mask], 0.f);
}

__global__ void z_combine(const float* __restrict__ agg, const float* __restrict__ b,
                          const float* __restrict__ rp, _Float16* __restrict__ o, long n){
  long i = (long)blockIdx.x*blockDim.x + threadIdx.x;
  long s = (long)gridDim.x*blockDim.x;
  for(; i < n; i += s) o[i] = (_Float16)(fmaxf(agg[i] + b[i & 511], 0.f) + rp[i]);
}

// ---------------- global attention pooling ----------------
__global__ void pool_logits(const float* __restrict__ zo, const float* __restrict__ gaW,
                            const float* __restrict__ gab, float* __restrict__ l, int n){
  int wid  = (blockIdx.x*blockDim.x + threadIdx.x) >> 5;
  int lane = threadIdx.x & 31;
  if(wid >= n) return;
  const float* zr = zo + (size_t)wid*128;
  float p = 0.f;
#pragma unroll
  for(int j = 0; j < 4; j++) p += zr[lane + j*32]*gaW[lane + j*32];
  p = wave_sum(p);
  if(lane == 0) l[wid] = p + gab[0];
}

__global__ void pool_max(const float* __restrict__ l, float* __restrict__ scal, int n){
  int i = blockIdx.x*blockDim.x + threadIdx.x;
  if(i < n) atomicMaxF(&scal[0], l[i]);
}

__global__ void pool_exp(const float* __restrict__ l, const float* __restrict__ scal,
                         float* __restrict__ e, float* __restrict__ sums, int n){
  int i = blockIdx.x*blockDim.x + threadIdx.x;
  if(i < n){ float v = expf(l[i] - scal[0]); e[i] = v; atomicAdd(&sums[0], v); }
}

__global__ void g_accum(const float* __restrict__ zo, const float* __restrict__ e,
                        float* __restrict__ g, int n){
  int c = threadIdx.x;                                // 128 threads = columns
  int chunk = (n + gridDim.x - 1)/gridDim.x;
  int n0 = blockIdx.x*chunk;
  int n1 = n0 + chunk; if(n1 > n) n1 = n;
  float p = 0.f;
  for(int i = n0; i < n1; i++) p += zo[(size_t)i*128 + c]*e[i];
  atomicAdd(&g[c], p);
}

__global__ void g_project(const float* __restrict__ g, const float* __restrict__ scal,
                          const float* __restrict__ gpW, const float* __restrict__ gpb,
                          float* __restrict__ gp){
  int c = threadIdx.x;
  float inv = 1.f/scal[1];
  float p = 0.f;
  for(int k = 0; k < 128; k++) p += g[k]*gpW[k*128 + c];
  gp[c] = 2.f*(p*inv + gpb[c]);
}

// z2 = zo + gp ; LayerNorm ; ReLU ; write f16 (wave per node)
__global__ void ln_relu(const float* __restrict__ zo, const float* __restrict__ gp,
                        const float* __restrict__ lng, const float* __restrict__ lnb,
                        _Float16* __restrict__ o, int n){
  int wid  = (blockIdx.x*blockDim.x + threadIdx.x) >> 5;
  int lane = threadIdx.x & 31;
  if(wid >= n) return;
  int c0 = lane*4;
  float4 zv = *(const float4*)(zo + (size_t)wid*128 + c0);
  float xx[4] = { zv.x + gp[c0], zv.y + gp[c0+1], zv.z + gp[c0+2], zv.w + gp[c0+3] };
  float s = 0.f, q = 0.f;
#pragma unroll
  for(int j = 0; j < 4; j++){ s += xx[j]; q += xx[j]*xx[j]; }
  s = wave_sum(s); q = wave_sum(q);
  float mu  = s*(1.f/128.f);
  float var = q*(1.f/128.f) - mu*mu;
  float rs  = rsqrtf(var + 1e-5f);
#pragma unroll
  for(int j = 0; j < 4; j++){
    float y = (xx[j] - mu)*rs*lng[c0+j] + lnb[c0+j];
    o[(size_t)wid*128 + c0 + j] = (_Float16)fmaxf(y, 0.f);
  }
}

__global__ void final_dot(const float* __restrict__ m1o, const float* __restrict__ m2W,
                          const float* __restrict__ m2b, float* __restrict__ out, int n){
  int wid  = (blockIdx.x*blockDim.x + threadIdx.x) >> 5;
  int lane = threadIdx.x & 31;
  if(wid >= n) return;
  const float* r = m1o + (size_t)wid*64;
  float p = r[lane]*m2W[lane] + r[lane+32]*m2W[lane+32];
  p = wave_sum(p);
  if(lane == 0) out[wid] = p + m2b[0];
}

// ---------------- driver ----------------
extern "C" void kernel_launch(void* const* d_in, const int* in_sizes, int n_in,
                              void* d_out, int out_size, void* d_ws, size_t ws_size,
                              hipStream_t stream)
{
  (void)in_sizes; (void)n_in; (void)out_size; (void)ws_size;
  const float* x    = (const float*)d_in[0];
  // d_in[1] edge_attr, d_in[4] eW, d_in[5] eb: do not affect output (edge_dim=None)
  const float* nW   = (const float*)d_in[2];
  const float* nb   = (const float*)d_in[3];
  const float* g1W  = (const float*)d_in[6];
  const float* g1as = (const float*)d_in[7];
  const float* g1ad = (const float*)d_in[8];
  const float* g1b  = (const float*)d_in[9];
  const float* g2W  = (const float*)d_in[10];
  const float* g2as = (const float*)d_in[11];
  const float* g2ad = (const float*)d_in[12];
  const float* g2b  = (const float*)d_in[13];
  const float* rW   = (const float*)d_in[14];
  const float* rb   = (const float*)d_in[15];
  const float* oW   = (const float*)d_in[16];
  const float* ob   = (const float*)d_in[17];
  const float* gaW  = (const float*)d_in[18];
  const float* gab  = (const float*)d_in[19];
  const float* gpW  = (const float*)d_in[20];
  const float* gpb  = (const float*)d_in[21];
  const float* lng  = (const float*)d_in[22];
  const float* lnb  = (const float*)d_in[23];
  const float* m1W  = (const float*)d_in[24];
  const float* m1b  = (const float*)d_in[25];
  const float* m2W  = (const float*)d_in[26];
  const float* m2b  = (const float*)d_in[27];
  const int*   ei   = (const int*)d_in[28];
  float* out = (float*)d_out;

  char* base = (char*)d_ws;
  size_t off = 0;
  auto alloc = [&](size_t bytes)->char*{
    off = (off + 255) & ~(size_t)255;
    char* p = base + off; off += bytes; return p;
  };
  _Float16* x16    = (_Float16*)alloc((size_t)NNODES*256*2);
  _Float16* nWT    = (_Float16*)alloc((size_t)256*128*2);   // [128 x 256]
  _Float16* g1WT   = (_Float16*)alloc((size_t)128*512*2);   // [512 x 128]
  _Float16* g2WT   = (_Float16*)alloc((size_t)512*512*2);   // [512 x 512]
  _Float16* rWT    = (_Float16*)alloc((size_t)128*512*2);   // [512 x 128]
  _Float16* oWT    = (_Float16*)alloc((size_t)512*128*2);   // [128 x 512]
  _Float16* m1WT   = (_Float16*)alloc((size_t)128*64*2);    // [64 x 128]
  _Float16* h16    = (_Float16*)alloc((size_t)NNODES*128*2); // h, later z2ln
  _Float16* t16    = (_Float16*)alloc((size_t)NNODES*512*2); // h1 f16, later z512 f16
  float* xh   = (float*)alloc((size_t)NNODES*512*4);          // xh1/xh2/rp/mlp1
  float* agg  = (float*)alloc((size_t)NNODES*512*4);
  float* a_s  = (float*)alloc((size_t)NNODES*4*4);
  float* a_d  = (float*)alloc((size_t)NNODES*4*4);
  float* amax = (float*)alloc((size_t)NNODES*4*4);
  float* den  = (float*)alloc((size_t)NNODES*4*4);
  float* exb  = (float*)alloc((size_t)TEDGES*4*4);
  float* zo   = (float*)alloc((size_t)NNODES*128*4);
  float* lg   = (float*)alloc((size_t)NNODES*4);
  float* ebuf = (float*)alloc((size_t)NNODES*4);
  float* gvec = (float*)alloc(128*4);
  float* gp   = (float*)alloc(128*4);
  float* scal = (float*)alloc(2*4);

  const float NEG_INF = -__builtin_inff();
  auto gs = [](long n){ return (int)((n + 255)/256); };

  // f16 staging: A-side plain convert, weights converted + transposed to [N x K]
  cvt_f32_f16<<<gs((long)NNODES*256),256,0,stream>>>(x, x16, (long)NNODES*256);
  cvt_t_f16<<<gs(256*128),256,0,stream>>>(nW,  nWT,  256, 128);
  cvt_t_f16<<<gs(128*512),256,0,stream>>>(g1W, g1WT, 128, 512);
  cvt_t_f16<<<gs(512*512),256,0,stream>>>(g2W, g2WT, 512, 512);
  cvt_t_f16<<<gs(128*512),256,0,stream>>>(rW,  rWT,  128, 512);
  cvt_t_f16<<<gs(512*128),256,0,stream>>>(oW,  oWT,  512, 128);
  cvt_t_f16<<<gs(128*64), 256,0,stream>>>(m1W, m1WT, 128, 64);

  const int MB = (NNODES + 127)/128;
  const int nodeBlocks = (NNODES + 7)/8;              // wave-per-node kernels
  const int edgeBlocks = (TEDGES + 255)/256;

  // h = relu(x @ nW + nb) -> f16
  wmma_gemm<<<dim3(MB,2),256,0,stream>>>(x16, nWT, nb, nullptr, h16, NNODES, 128, 256, 1);
  // xh1 = h @ g1W
  wmma_gemm<<<dim3(MB,8),256,0,stream>>>(h16, g1WT, nullptr, xh, nullptr, NNODES, 512, 128, 0);

  for(int layer = 0; layer < 2; ++layer){
    const float* as_w = layer == 0 ? g1as : g2as;
    const float* ad_w = layer == 0 ? g1ad : g2ad;
    gat_scores<<<nodeBlocks,256,0,stream>>>(xh, as_w, ad_w, a_s, a_d, NNODES);
    fill_f32<<<gs(NNODES*4),256,0,stream>>>(amax, NEG_INF, NNODES*4);
    fill_f32<<<gs(NNODES*4),256,0,stream>>>(den, 0.f, NNODES*4);
    fill_f32<<<gs((long)NNODES*512),256,0,stream>>>(agg, 0.f, (long)NNODES*512);
    edge_max   <<<edgeBlocks,256,0,stream>>>(ei, a_s, a_d, amax);
    edge_expsum<<<edgeBlocks,256,0,stream>>>(ei, a_s, a_d, amax, exb, den);
    edge_aggregate<<<TEDGES,128,0,stream>>>(ei, xh, exb, den, agg);
    if(layer == 0){
      // h1 = relu(agg + g1b) -> f16 ; xh2 = h1 @ g2W
      bias_relu_cvt<<<gs((long)NNODES*512),256,0,stream>>>(agg, g1b, t16, (long)NNODES*512, 511);
      wmma_gemm<<<dim3(MB,8),256,0,stream>>>(t16, g2WT, nullptr, xh, nullptr, NNODES, 512, 512, 0);
    }
  }

  // rp = h @ rW + rb (overwrites xh) ; z512 = relu(agg + g2b) + rp -> f16
  wmma_gemm<<<dim3(MB,8),256,0,stream>>>(h16, rWT, rb, xh, nullptr, NNODES, 512, 128, 0);
  z_combine<<<gs((long)NNODES*512),256,0,stream>>>(agg, g2b, xh, t16, (long)NNODES*512);
  // zo = z512 @ oW + ob
  wmma_gemm<<<dim3(MB,2),256,0,stream>>>(t16, oWT, ob, zo, nullptr, NNODES, 128, 512, 0);

  // global softmax-attention pooling
  pool_logits<<<nodeBlocks,256,0,stream>>>(zo, gaW, gab, lg, NNODES);
  fill_f32<<<1,32,0,stream>>>(scal,   NEG_INF, 1);
  fill_f32<<<1,32,0,stream>>>(scal+1, 0.f,     1);
  pool_max<<<gs(NNODES),256,0,stream>>>(lg, scal, NNODES);
  pool_exp<<<gs(NNODES),256,0,stream>>>(lg, scal, ebuf, scal+1, NNODES);
  fill_f32<<<1,128,0,stream>>>(gvec, 0.f, 128);
  g_accum<<<160,128,0,stream>>>(zo, ebuf, gvec, NNODES);
  g_project<<<1,128,0,stream>>>(gvec, scal, gpW, gpb, gp);

  // z2 = zo + gp -> LN -> relu -> f16 (reuse h16)
  ln_relu<<<nodeBlocks,256,0,stream>>>(zo, gp, lng, lnb, h16, NNODES);
  // mlp1 = relu(z2 @ m1W + m1b) (reuse xh)
  wmma_gemm<<<dim3(MB,1),256,0,stream>>>(h16, m1WT, m1b, xh, nullptr, NNODES, 64, 128, 1);
  // out = mlp1 @ m2W + m2b
  final_dot<<<nodeBlocks,256,0,stream>>>(xh, m2W, m2b, out, NNODES);
}
